// ImplicitLatentAutoEncoder_19439021981843
// MI455X (gfx1250) — matrix operand order repaired and checked by reading, not compiled, not run on hardware
//
#include <hip/hip_runtime.h>

typedef _Float16 f16;
typedef __attribute__((ext_vector_type(8)))  _Float16 v8h;
typedef __attribute__((ext_vector_type(16))) _Float16 v16h;
typedef __attribute__((ext_vector_type(8)))  float    v8f;

#define DEV static __device__ __forceinline__

// ---------------------------------------------------------------------------
// WMMA 16x16x32 f16 fragment loaders (wave32, layouts per CDNA5 ISA 7.12.2)
// ---------------------------------------------------------------------------
// A (16x32, MxK) from row-major LDS: lanes 0-15 -> M=lane, K 0..7 & 16..23;
// lanes 16-31 -> M=lane-16, K 8..15 & 24..31.  Two 16B LDS loads per lane.
DEV v16h load_a_frag(const f16* base, int stride, int k0, int lane) {
  const int r  = lane & 15;
  const int kb = k0 + ((lane & 16) ? 8 : 0);
  const f16* p = base + r * stride + kb;
  v8h lo = *(const v8h*)(p);
  v8h hi = *(const v8h*)(p + 16);
  return __builtin_shufflevector(lo, hi, 0, 1, 2, 3, 4, 5, 6, 7,
                                 8, 9, 10, 11, 12, 13, 14, 15);
}

// B fragments are pre-swizzled in the prep kernel: fragment (kt,nt) is stored
// as 32 lanes x 16 halves contiguous -> one aligned 32B load per lane.
DEV v16h load_b_frag(const f16* Wsw, int frag, int lane) {
  return *(const v16h*)(Wsw + frag * 512 + lane * 16);
}

// One dense layer: out[16][N] = relu(in[16][K] @ W[K][N] + bias), all LDS f16.
// Compile-time dims so A fragments are hoisted into registers (loaded once,
// reused across all n-tiles).  Accumulator starts at inline-0 (free C operand);
// bias+relu are applied as independent post-ops that co-execute with the next
// tile's WMMA stream.  OUTS >= N; pad columns [N, OUTS) are zeroed.
template <int K, int N, int INS, int OUTS>
DEV void mlp_layer(const f16* in, const f16* Wsw, const float* bias,
                   f16* out, int lane) {
  constexpr int ktiles = K / 32;
  constexpr int ntiles = N / 16;
  const int rbase = (lane & 16) ? 8 : 0;
  const int cl    = lane & 15;

  v16h a[ktiles];
#pragma unroll
  for (int kt = 0; kt < ktiles; ++kt) a[kt] = load_a_frag(in, INS, kt * 32, lane);

#pragma unroll
  for (int nt = 0; nt < ntiles; ++nt) {
    v8f acc = {};
#pragma unroll
    for (int kt = 0; kt < ktiles; ++kt) {
      v16h b = load_b_frag(Wsw, kt * ntiles + nt, lane);
      acc = __builtin_amdgcn_wmma_f32_16x16x32_f16(false, a[kt], false, b,
                                                   (short)0, acc, false, false);
    }
    const int   col = nt * 16 + cl;
    const float bv  = bias[col];
#pragma unroll
    for (int i = 0; i < 8; ++i) {
      float v = acc[i] + bv;
      v = v > 0.0f ? v : 0.0f;
      out[(rbase + i) * OUTS + col] = (f16)v;
    }
  }
  if (OUTS > N) {
#pragma unroll
    for (int pc = N; pc < OUTS; pc += 16) {
#pragma unroll
      for (int i = 0; i < 8; ++i) out[(rbase + i) * OUTS + pc + cl] = (f16)0.0f;
    }
  }
}

// Final edge layer (K=128 -> N=128): relu, exclude diagonal row, accumulate
// per-object row sums.  A frags hoisted; inline-0 accumulator.
DEV void edge_final_accum(const f16* in /*stride 128*/, const f16* Wsw,
                          const float* bias, int diag_row, int lane,
                          float* sums /*[8]*/) {
  const int rbase = (lane & 16) ? 8 : 0;
  const int cl    = lane & 15;
  v16h a[4];
#pragma unroll
  for (int kt = 0; kt < 4; ++kt) a[kt] = load_a_frag(in, 128, kt * 32, lane);
#pragma unroll
  for (int nt = 0; nt < 8; ++nt) {
    v8f acc = {};
#pragma unroll
    for (int kt = 0; kt < 4; ++kt) {
      v16h b = load_b_frag(Wsw, kt * 8 + nt, lane);
      acc = __builtin_amdgcn_wmma_f32_16x16x32_f16(false, a[kt], false, b,
                                                   (short)0, acc, false, false);
    }
    const float bv = bias[nt * 16 + cl];
    float s = 0.0f;
#pragma unroll
    for (int i = 0; i < 8; ++i) {
      float v = acc[i] + bv;
      v = v > 0.0f ? v : 0.0f;
      if ((rbase + i) != diag_row) s += v;
    }
    s += __shfl_xor(s, 16, 32);  // combine lane-half partial sums (M halves)
    sums[nt] += s;
  }
}

// ---------------------------------------------------------------------------
// Prep kernels
// ---------------------------------------------------------------------------
// Convert W (K x N f32, row-major) into f16 B-fragment-swizzled layout with K
// zero-padded to Kpad.  dst[frag*512 + lane*16 + e], frag=(kt*ntiles+nt).
__global__ void convert_w_kernel(const float* __restrict__ src, f16* __restrict__ dst,
                                 int K, int N, int Kpad) {
  int i = blockIdx.x * blockDim.x + threadIdx.x;
  if (i >= Kpad * N) return;
  int ntiles = N >> 4;
  int f = i >> 9, r = i & 511;
  int lane = r >> 4, e = r & 15;
  int kt = f / ntiles, nt = f - kt * ntiles;
  int k = kt * 32 + ((lane & 16) ? 8 : 0) + (e < 8 ? e : e + 8);
  int c = nt * 16 + (lane & 15);
  dst[i] = (k < K) ? (f16)src[k * N + c] : (f16)0.0f;
}

// nodes[b][o][s*4+e] = x[b][s][o][e]   (B=512, SEQ=8, O=32, EXP=4)
__global__ void build_nodes_kernel(const float* __restrict__ x, f16* __restrict__ nodes) {
  int i = blockIdx.x * blockDim.x + threadIdx.x;
  if (i >= 512 * 32 * 32) return;
  int e = i & 3, s = (i >> 2) & 7, o = (i >> 5) & 31, b = i >> 10;
  nodes[i] = (f16)x[((b * 8 + s) * 32 + o) * 4 + e];
}

// ---------------------------------------------------------------------------
// Stage 1: edge_enc pairwise MLP + neighbor mean.  1 block = 1 batch.
// ---------------------------------------------------------------------------
__global__ void __launch_bounds__(256)
edge_enc_kernel(const f16* __restrict__ Wall,
                const float* __restrict__ b0, const float* __restrict__ b1,
                const float* __restrict__ b2, const float* __restrict__ b3,
                const f16* __restrict__ nodes_g, float* __restrict__ edge_mean) {
  extern __shared__ __align__(16) char smem[];
  f16*   Wl      = (f16*)smem;           // 57344 halves (swizzled)
  f16*   nodes   = Wl + 57344;           // 32x32
  f16*   scratch = nodes + 1024;         // 8 waves * 2 * 2048 halves
  float* bl      = (float*)(scratch + 8 * 4096);  // 4*128 biases

  const int tid = threadIdx.x, lane = tid & 31, wave = tid >> 5;
  const int b = blockIdx.x;

  for (int i = tid; i < 57344 / 8; i += 256)
    ((uint4*)Wl)[i] = ((const uint4*)Wall)[i];
  for (int i = tid; i < 128; i += 256)
    ((uint4*)nodes)[i] = ((const uint4*)(nodes_g + (size_t)b * 1024))[i];
  if (tid < 128) {
    bl[tid] = b0[tid]; bl[128 + tid] = b1[tid];
    bl[256 + tid] = b2[tid]; bl[384 + tid] = b3[tid];
  }
  __syncthreads();

  f16* bufA = scratch + wave * 4096;
  f16* bufB = bufA + 2048;
  const f16 *W0 = Wl, *W1 = Wl + 8192, *W2 = Wl + 24576, *W3 = Wl + 40960;

  for (int j = 0; j < 4; ++j) {
    const int o = wave * 4 + j;
    float sums[8];
#pragma unroll
    for (int t = 0; t < 8; ++t) sums[t] = 0.0f;
    for (int half = 0; half < 2; ++half) {
      const int n0 = half * 16;
      // in = concat(nodes[o], nodes[n0+r]) : 16x64, 8-half vector chunks
      for (int idx = lane; idx < 128; idx += 32) {
        int r = idx >> 3, c8 = (idx & 7) * 8;
        const f16* sp = (c8 < 32) ? (nodes + o * 32 + c8)
                                  : (nodes + (n0 + r) * 32 + (c8 - 32));
        *(v8h*)(bufA + r * 64 + c8) = *(const v8h*)sp;
      }
      __syncthreads();
      mlp_layer<64, 128, 64, 128>(bufA, W0, bl, bufB, lane);         __syncthreads();
      mlp_layer<128, 128, 128, 128>(bufB, W1, bl + 128, bufA, lane); __syncthreads();
      mlp_layer<128, 128, 128, 128>(bufA, W2, bl + 256, bufB, lane); __syncthreads();
      const int diag = (o >= n0 && o < n0 + 16) ? (o - n0) : -1;
      edge_final_accum(bufB, W3, bl + 384, diag, lane, sums);
      __syncthreads();
    }
    if (lane < 16) {
      float* d = edge_mean + ((size_t)b * 32 + o) * 128;
#pragma unroll
      for (int nt = 0; nt < 8; ++nt) d[nt * 16 + lane] = sums[nt] * (1.0f / 31.0f);
    }
  }
}

// ---------------------------------------------------------------------------
// Stage 2: node_enc.  1 wave = one 16-row tile of the 16384 nodes.
// ---------------------------------------------------------------------------
__global__ void __launch_bounds__(256)
node_enc_kernel(const f16* __restrict__ Wall,
                const float* b0, const float* b1, const float* b2,
                const float* b3, const float* b4,
                const f16* __restrict__ nodes_g, const float* __restrict__ em,
                f16* __restrict__ z) {
  extern __shared__ __align__(16) char smem[];
  f16*   Wl      = (f16*)smem;           // 31744 halves (swizzled)
  f16*   scratch = Wl + 31744;           // 8 waves * 2 * 2560 halves
  float* bl      = (float*)(scratch + 8 * 5120);  // 256 biases

  const int tid = threadIdx.x, lane = tid & 31, wave = tid >> 5;
  const f16* Wsrc = Wall + 57344;
  for (int i = tid; i < 31744 / 8; i += 256)
    ((uint4*)Wl)[i] = ((const uint4*)Wsrc)[i];
  if (tid < 128) bl[tid] = b0[tid];
  if (tid < 64)  bl[128 + tid] = b1[tid];
  if (tid < 32)  bl[192 + tid] = b2[tid];
  if (tid < 16)  { bl[224 + tid] = b3[tid]; bl[240 + tid] = b4[tid]; }
  __syncthreads();

  f16* bufA = scratch + wave * 5120;
  f16* bufB = bufA + 2560;
  const f16 *W0 = Wl, *W1 = Wl + 20480, *W2 = Wl + 28672, *W3 = Wl + 30720,
            *W4 = Wl + 31232;
  const int rt = blockIdx.x * 8 + wave;

  // in = concat(nodes, edge_mean) : 16x160, 20 chunks of 8 per row
  for (int idx = lane; idx < 320; idx += 32) {
    int r = idx / 20, c8 = (idx % 20) * 8;
    size_t g = (size_t)rt * 16 + r;
    v8h v;
    if (c8 < 32) {
      v = *(const v8h*)(nodes_g + g * 32 + c8);
    } else {
      const float* ep = em + g * 128 + (c8 - 32);
#pragma unroll
      for (int e = 0; e < 8; ++e) v[e] = (f16)ep[e];
    }
    *(v8h*)(bufA + r * 160 + c8) = v;
  }
  __syncthreads();
  mlp_layer<160, 128, 160, 128>(bufA, W0, bl, bufB, lane);       __syncthreads();
  mlp_layer<128, 64, 128, 64>(bufB, W1, bl + 128, bufA, lane);   __syncthreads();
  mlp_layer<64, 32, 64, 32>(bufA, W2, bl + 192, bufB, lane);     __syncthreads();
  mlp_layer<32, 16, 32, 32>(bufB, W3, bl + 224, bufA, lane);     __syncthreads(); // pad K->32
  mlp_layer<32, 16, 32, 16>(bufA, W4, bl + 240, bufB, lane);     __syncthreads();

  {  // write z (16x16 f16) as 8-half chunks
    int r = lane >> 1, c8 = (lane & 1) * 8;
    *(v8h*)(z + ((size_t)rt * 16 + r) * 16 + c8) = *(const v8h*)(bufB + r * 16 + c8);
  }
}

// ---------------------------------------------------------------------------
// Stage 3: edge_dec pairwise MLP + neighbor mean.  1 block = 1 batch.
// ---------------------------------------------------------------------------
__global__ void __launch_bounds__(256)
edge_dec_kernel(const f16* __restrict__ Wall,
                const float* b0, const float* b1, const float* b2, const float* b3,
                const f16* __restrict__ z_g, float* __restrict__ emd) {
  extern __shared__ __align__(16) char smem[];
  f16*   Wl      = (f16*)smem;           // 27648 halves (swizzled)
  f16*   zsh     = Wl + 27648;           // 32x16
  f16*   scratch = zsh + 512;            // 8 waves * 2 * 2048 halves
  float* bl      = (float*)(scratch + 8 * 4096);  // 352 biases

  const int tid = threadIdx.x, lane = tid & 31, wave = tid >> 5;
  const int b = blockIdx.x;
  const f16* Wsrc = Wall + 89088;
  for (int i = tid; i < 27648 / 8; i += 256)
    ((uint4*)Wl)[i] = ((const uint4*)Wsrc)[i];
  for (int i = tid; i < 64; i += 256)
    ((uint4*)zsh)[i] = ((const uint4*)(z_g + (size_t)b * 512))[i];
  if (tid < 32)  bl[tid] = b0[tid];
  if (tid < 64)  bl[32 + tid] = b1[tid];
  if (tid < 128) { bl[96 + tid] = b2[tid]; bl[224 + tid] = b3[tid]; }
  __syncthreads();

  f16* bufA = scratch + wave * 4096;
  f16* bufB = bufA + 2048;
  const f16 *W0 = Wl, *W1 = Wl + 1024, *W2 = Wl + 3072, *W3 = Wl + 11264;

  for (int j = 0; j < 4; ++j) {
    const int o = wave * 4 + j;
    float sums[8];
#pragma unroll
    for (int t = 0; t < 8; ++t) sums[t] = 0.0f;
    for (int half = 0; half < 2; ++half) {
      const int n0 = half * 16;
      // in = concat(z[o], z[n0+r]) : 16x32, 4 chunks of 8 per row
      for (int idx = lane; idx < 64; idx += 32) {
        int r = idx >> 2, c8 = (idx & 3) * 8;
        const f16* sp = (c8 < 16) ? (zsh + o * 16 + c8)
                                  : (zsh + (n0 + r) * 16 + (c8 - 16));
        *(v8h*)(bufA + r * 32 + c8) = *(const v8h*)sp;
      }
      __syncthreads();
      mlp_layer<32, 32, 32, 32>(bufA, W0, bl, bufB, lane);        __syncthreads();
      mlp_layer<32, 64, 32, 64>(bufB, W1, bl + 32, bufA, lane);   __syncthreads();
      mlp_layer<64, 128, 64, 128>(bufA, W2, bl + 96, bufB, lane); __syncthreads();
      const int diag = (o >= n0 && o < n0 + 16) ? (o - n0) : -1;
      edge_final_accum(bufB, W3, bl + 224, diag, lane, sums);
      __syncthreads();
    }
    if (lane < 16) {
      float* d = emd + ((size_t)b * 32 + o) * 128;
#pragma unroll
      for (int nt = 0; nt < 8; ++nt) d[nt * 16 + lane] = sums[nt] * (1.0f / 31.0f);
    }
  }
}

// ---------------------------------------------------------------------------
// Stage 4: node_dec + output transpose (b,o,s,e) -> (b,s,o,e).
// ---------------------------------------------------------------------------
__global__ void __launch_bounds__(256)
node_dec_kernel(const f16* __restrict__ Wall,
                const float* b0, const float* b1, const float* b2,
                const float* b3, const float* b4,
                const f16* __restrict__ z_g, const float* __restrict__ emd,
                float* __restrict__ out) {
  extern __shared__ __align__(16) char smem[];
  f16*   Wl      = (f16*)smem;           // 48128 halves (swizzled)
  f16*   scratch = Wl + 48128;           // 8 waves * 2 * 2560 halves
  float* bl      = (float*)(scratch + 8 * 5120);  // 384 biases

  const int tid = threadIdx.x, lane = tid & 31, wave = tid >> 5;
  const f16* Wsrc = Wall + 116736;
  for (int i = tid; i < 48128 / 8; i += 256)
    ((uint4*)Wl)[i] = ((const uint4*)Wsrc)[i];
  if (tid < 128) { bl[tid] = b0[tid]; bl[128 + tid] = b1[tid]; }
  if (tid < 64)  bl[256 + tid] = b2[tid];
  if (tid < 32)  { bl[320 + tid] = b3[tid]; bl[352 + tid] = b4[tid]; }
  __syncthreads();

  f16* bufA = scratch + wave * 5120;
  f16* bufB = bufA + 2560;
  const f16 *W0 = Wl, *W1 = Wl + 20480, *W2 = Wl + 36864, *W3 = Wl + 45056,
            *W4 = Wl + 47104;
  const int rt = blockIdx.x * 8 + wave;

  // in = concat(z, emd) padded 144->160 : 16x160
  for (int idx = lane; idx < 320; idx += 32) {
    int r = idx / 20, c8 = (idx % 20) * 8;
    size_t g = (size_t)rt * 16 + r;
    v8h v;
    if (c8 < 16) {
      v = *(const v8h*)(z_g + g * 16 + c8);
    } else if (c8 < 144) {
      const float* ep = emd + g * 128 + (c8 - 16);
#pragma unroll
      for (int e = 0; e < 8; ++e) v[e] = (f16)ep[e];
    } else {
#pragma unroll
      for (int e = 0; e < 8; ++e) v[e] = (f16)0.0f;
    }
    *(v8h*)(bufA + r * 160 + c8) = v;
  }
  __syncthreads();
  mlp_layer<160, 128, 160, 128>(bufA, W0, bl, bufB, lane);        __syncthreads();
  mlp_layer<128, 128, 128, 128>(bufB, W1, bl + 128, bufA, lane);  __syncthreads();
  mlp_layer<128, 64, 128, 64>(bufA, W2, bl + 256, bufB, lane);    __syncthreads();
  mlp_layer<64, 32, 64, 32>(bufB, W3, bl + 320, bufA, lane);      __syncthreads();
  mlp_layer<32, 32, 32, 32>(bufA, W4, bl + 352, bufB, lane);      __syncthreads();

  // scatter with transpose; 4 consecutive e's form one float4 store
  for (int idx = lane; idx < 128; idx += 32) {
    int r = idx >> 3, s = idx & 7;
    size_t g = (size_t)rt * 16 + r;
    size_t bb = g >> 5, o = g & 31;
    float4 v;
    v.x = (float)bufB[r * 32 + s * 4 + 0];
    v.y = (float)bufB[r * 32 + s * 4 + 1];
    v.z = (float)bufB[r * 32 + s * 4 + 2];
    v.w = (float)bufB[r * 32 + s * 4 + 3];
    *(float4*)(out + ((bb * 8 + s) * 32 + o) * 4) = v;
  }
}

// ---------------------------------------------------------------------------
// Host launch
// ---------------------------------------------------------------------------
extern "C" void kernel_launch(void* const* d_in, const int* in_sizes, int n_in,
                              void* d_out, int out_size, void* d_ws, size_t ws_size,
                              hipStream_t stream) {
  // Workspace layout (bytes)
  f16*   Wall  = (f16*)d_ws;                                  // 164864 halves
  f16*   nodes = (f16*)((char*)d_ws + 329728);                // (512,32,32) f16
  f16*   z     = (f16*)((char*)d_ws + 1378304);               // (512,32,16) f16
  float* em    = (float*)((char*)d_ws + 1902592);             // (512,32,128) f32
  float* emd   = (float*)((char*)d_ws + 10291200);            // (512,32,128) f32

  // Logical tensor order: EE(W,b)x4, NE x5, ED x4, ND x5, then x.
  // Case A: dict insertion order -> x first.  Case B: sorted-key flatten.
  static const int idxA[37] = { 1,2,3,4,5,6,7,8,
                                9,10,11,12,13,14,15,16,17,18,
                                19,20,21,22,23,24,25,26,
                                27,28,29,30,31,32,33,34,35,36,
                                0 };
  static const int idxB[37] = { 8,9,10,11,12,13,14,15,
                                26,27,28,29,30,31,32,33,34,35,
                                0,1,2,3,4,5,6,7,
                                16,17,18,19,20,21,22,23,24,25,
                                36 };
  const int* ix = (in_sizes[0] == 512 * 8 * 32 * 4) ? idxA : idxB;

  struct WSpec { int li, K, N, Kpad, off; };
  static const WSpec WS[18] = {
    { 0,  64, 128,  64,      0}, { 2, 128, 128, 128,   8192},
    { 4, 128, 128, 128,  24576}, { 6, 128, 128, 128,  40960},
    { 8, 160, 128, 160,  57344}, {10, 128,  64, 128,  77824},
    {12,  64,  32,  64,  86016}, {14,  32,  16,  32,  88064},
    {16,  16,  16,  32,  88576},
    {18,  32,  32,  32,  89088}, {20,  32,  64,  32,  90112},
    {22,  64, 128,  64,  92160}, {24, 128, 128, 128, 100352},
    {26, 144, 128, 160, 116736}, {28, 128, 128, 128, 137216},
    {30, 128,  64, 128, 153600}, {32,  64,  32,  64, 161792},
    {34,  32,  32,  32, 163840},
  };
  for (int i = 0; i < 18; ++i) {
    const WSpec& s = WS[i];
    int total = s.Kpad * s.N;
    convert_w_kernel<<<(total + 255) / 256, 256, 0, stream>>>(
        (const float*)d_in[ix[s.li]], Wall + s.off, s.K, s.N, s.Kpad);
  }
  build_nodes_kernel<<<(512 * 32 * 32 + 255) / 256, 256, 0, stream>>>(
      (const float*)d_in[ix[36]], nodes);

  const float* eb[4] = { (const float*)d_in[ix[1]],  (const float*)d_in[ix[3]],
                         (const float*)d_in[ix[5]],  (const float*)d_in[ix[7]] };
  const float* nb[5] = { (const float*)d_in[ix[9]],  (const float*)d_in[ix[11]],
                         (const float*)d_in[ix[13]], (const float*)d_in[ix[15]],
                         (const float*)d_in[ix[17]] };
  const float* db[4] = { (const float*)d_in[ix[19]], (const float*)d_in[ix[21]],
                         (const float*)d_in[ix[23]], (const float*)d_in[ix[25]] };
  const float* ob[5] = { (const float*)d_in[ix[27]], (const float*)d_in[ix[29]],
                         (const float*)d_in[ix[31]], (const float*)d_in[ix[33]],
                         (const float*)d_in[ix[35]] };

  // Allow >64KB dynamic LDS (CDNA5 WGP: up to 320KB per workgroup).
  hipFuncSetAttribute(reinterpret_cast<const void*>(edge_enc_kernel),
                      hipFuncAttributeMaxDynamicSharedMemorySize, 184320);
  hipFuncSetAttribute(reinterpret_cast<const void*>(node_enc_kernel),
                      hipFuncAttributeMaxDynamicSharedMemorySize, 146432);
  hipFuncSetAttribute(reinterpret_cast<const void*>(edge_dec_kernel),
                      hipFuncAttributeMaxDynamicSharedMemorySize, 123264);
  hipFuncSetAttribute(reinterpret_cast<const void*>(node_dec_kernel),
                      hipFuncAttributeMaxDynamicSharedMemorySize, 179712);

  edge_enc_kernel<<<512, 256, 184320, stream>>>(Wall, eb[0], eb[1], eb[2], eb[3],
                                                nodes, em);
  node_enc_kernel<<<128, 256, 146432, stream>>>(Wall, nb[0], nb[1], nb[2], nb[3], nb[4],
                                                nodes, em, z);
  edge_dec_kernel<<<512, 256, 123264, stream>>>(Wall, db[0], db[1], db[2], db[3],
                                                z, emd);
  node_dec_kernel<<<128, 256, 179712, stream>>>(Wall, ob[0], ob[1], ob[2], ob[3], ob[4],
                                                z, emd, (float*)d_out);
}
